// GeometricSparseAttention_52888227283525
// MI455X (gfx1250) — matrix operand order repaired and checked by reading, not compiled, not run on hardware
//
#include <hip/hip_runtime.h>

typedef __attribute__((ext_vector_type(16))) _Float16 v16h;
typedef __attribute__((ext_vector_type(8)))  _Float16 v8h;
typedef __attribute__((ext_vector_type(8)))  float    v8f;

#define SEQ   1024
#define DMODEL 512
#define NHEAD 8
#define HDIM  64
#define MROWS 2048     // B*S
#define NBINS 512
#define KSPARSE 307    // int(1024*0.3)
#define PI_F 3.14159265358979f

// ---------------------------------------------------------------------------------------
// Fragment-native LDS staging (CDNA5 ISA 7.12.2 wave32 layouts).
// A fragment (16x32 f16): lane = r + 16*((k>>3)&1), elem = (k&7) | ((k>>4)<<3)
// B fragment (32x16 f16): lane = n + 16*((k>>3)&1), elem = (k&7) | ((k>>4)<<3)
// 8 contiguous global K-elements (8-aligned) land in 8 contiguous halves of ONE lane's
// v16h slot -> staging is one 16B global load + one 16B LDS store per chunk.
// ---------------------------------------------------------------------------------------
__device__ __forceinline__ _Float16* a_dst(v16h Afrag[2][4][32], int r, int c0) {
  int ks = c0 >> 5, c = c0 & 31;
  int lane = (r & 15) + (((c >> 3) & 1) << 4);
  int e0 = ((c >> 4) & 1) << 3;
  return (_Float16*)&Afrag[ks][r >> 4][lane] + e0;
}
__device__ __forceinline__ _Float16* b_dst(v16h Bfrag[2][2][32], int n, int k0l) {
  int ks = k0l >> 5, kc = k0l & 31;
  int lane = (n & 15) + (((kc >> 3) & 1) << 4);
  return (_Float16*)&Bfrag[ks][n >> 4][lane] + ((((kc >> 4) & 1) << 3) | (kc & 7));
}
__device__ __forceinline__ v8h cvt_f32x8(const float* p) {
  float4 f0 = *(const float4*)p;
  float4 f1 = *(const float4*)(p + 4);
  v8h h;
  h[0] = (_Float16)f0.x; h[1] = (_Float16)f0.y; h[2] = (_Float16)f0.z; h[3] = (_Float16)f0.w;
  h[4] = (_Float16)f1.x; h[5] = (_Float16)f1.y; h[6] = (_Float16)f1.z; h[7] = (_Float16)f1.w;
  return h;
}

__device__ __forceinline__ float wave_max(float v) {
#pragma unroll
  for (int o = 16; o; o >>= 1) v = fmaxf(v, __shfl_xor(v, o, 32));
  return v;
}
__device__ __forceinline__ float wave_sum(float v) {
#pragma unroll
  for (int o = 16; o; o >>= 1) v += __shfl_xor(v, o, 32);
  return v;
}
__device__ __forceinline__ float wave_min(float v) {
#pragma unroll
  for (int o = 16; o; o >>= 1) v = fminf(v, __shfl_xor(v, o, 32));
  return v;
}
__device__ __forceinline__ int wave_isum(int v) {
#pragma unroll
  for (int o = 16; o; o >>= 1) v += __shfl_xor(v, o, 32);
  return v;
}

// ---------------- K1/K8: y = A @ W^T + bias (W is [N,K] torch layout) -------------------
// 256 threads = 8 waves; block tile 64(m) x 32(n); wave (w>>1, w&1) -> one 16x16 tile.
// BK = 64 -> two back-to-back v_wmma per barrier; K-loop unrolled x2.
// DESTMODE 0: f16 [B,H,S,hd];  DESTMODE 1: f32 row-major [M,N];  DESTMODE 2: f16 [B,H,hd,S]
template <int AISHALF, int DESTMODE>
__global__ void gemm_wt_kernel(const void* __restrict__ Aptr,
                               const float* __restrict__ W, const float* __restrict__ bias,
                               void* __restrict__ Dptr, int K, int N) {
  __shared__ v16h Afrag[2][4][32];   // [kstep][wm][lane]
  __shared__ v16h Bfrag[2][2][32];   // [kstep][wn][lane]
  int tid = threadIdx.x;
  int lane = tid & 31, wave = tid >> 5;
  int wm = wave >> 1, wn = wave & 1;
  int m0 = blockIdx.x * 64, n0 = blockIdx.y * 32;
  v8f acc = {};
#pragma unroll 2
  for (int k0 = 0; k0 < K; k0 += 64) {
    // stage A (64 rows x 64 k): 512 16B-chunks, 2 per thread
#pragma unroll
    for (int it = 0; it < 2; ++it) {
      int cid = tid + it * 256;
      int r = cid >> 3, c0 = (cid & 7) << 3;
      v8h h;
      if (AISHALF) h = *(const v8h*)&((const _Float16*)Aptr)[(m0 + r) * K + k0 + c0];
      else         h = cvt_f32x8(&((const float*)Aptr)[(m0 + r) * K + k0 + c0]);
      *(v8h*)a_dst(Afrag, r, c0) = h;
    }
    // stage B (32 n x 64 k): 256 chunks, 1 per thread
    {
      int n = tid >> 3, k0l = (tid & 7) << 3;
      v8h h = cvt_f32x8(&W[(n0 + n) * K + k0 + k0l]);
      *(v8h*)b_dst(Bfrag, n, k0l) = h;
    }
    if (k0 + 64 < K) __builtin_prefetch(&W[n0 * K + k0 + 64], 0, 1);
    __syncthreads();
    v16h a0 = Afrag[0][wm][lane], b0 = Bfrag[0][wn][lane];
    v16h a1 = Afrag[1][wm][lane], b1 = Bfrag[1][wn][lane];
    acc = __builtin_amdgcn_wmma_f32_16x16x32_f16(false, a0, false, b0, (short)0, acc, false, false);
    acc = __builtin_amdgcn_wmma_f32_16x16x32_f16(false, a1, false, b1, (short)0, acc, false, false);
    __syncthreads();
  }
  int nloc = lane & 15, mhalf = lane >> 4;
#pragma unroll
  for (int i = 0; i < 8; ++i) {
    int m = m0 + wm * 16 + mhalf * 8 + i;   // C/D layout: VGPR i -> row i + 8*half
    int n = n0 + wn * 16 + nloc;
    float val = acc[i] + bias[n];
    int bb = m >> 10, s = m & 1023;
    int h = n >> 6, d = n & 63;
    if (DESTMODE == 0) {
      ((_Float16*)Dptr)[(((bb * NHEAD + h) << 10) + s) * HDIM + d] = (_Float16)val;
    } else if (DESTMODE == 2) {
      ((_Float16*)Dptr)[((bb * NHEAD + h) * HDIM + d) * SEQ + s] = (_Float16)val;  // transposed V
    } else {
      ((float*)Dptr)[m * N + n] = val;
    }
  }
}

// ---------------- K2: row L2 norms of q and k -------------------------------------------
__global__ void norms_kernel(const _Float16* __restrict__ q, const _Float16* __restrict__ k,
                             float* __restrict__ qn, float* __restrict__ kn) {
  int lane = threadIdx.x & 31, wave = threadIdx.x >> 5;
  int row = blockIdx.x * 8 + wave;
  float sq = 0.f, sk = 0.f;
#pragma unroll
  for (int t = 0; t < 2; ++t) {
    float a = (float)q[row * HDIM + lane + t * 32];
    float b = (float)k[row * HDIM + lane + t * 32];
    sq += a * a; sk += b * b;
  }
  sq = wave_sum(sq); sk = wave_sum(sk);
  if (lane == 0) { qn[row] = sqrtf(sq); kn[row] = sqrtf(sk); }
}

// ---------------- K3: raw QK^T via WMMA; MODE 0 = geo histogram, MODE 1 = scores+gmask ---
// block tile 64(i) x 32(j); K = hd = 64 -> straight-line, 2 WMMAs, no K-loop
template <int MODE>
__global__ void scores_kernel(const _Float16* __restrict__ q, const _Float16* __restrict__ k,
                              const float* __restrict__ qn, const float* __restrict__ kn,
                              float* __restrict__ scores, unsigned char* __restrict__ gmask,
                              unsigned int* __restrict__ histb, const float* __restrict__ thrb) {
  __shared__ v16h Afrag[2][4][32];
  __shared__ v16h Bfrag[2][2][32];
  __shared__ unsigned int lhist[NBINS];
  int tid = threadIdx.x;
  int lane = tid & 31, wave = tid >> 5;
  int wm = wave >> 1, wn = wave & 1;
  int i0 = blockIdx.x * 64, j0 = blockIdx.y * 32;
  if (MODE == 0) {
    for (int t = tid; t < NBINS; t += 256) lhist[t] = 0u;
  }
  // stage q tile (64 x 64): vectorized
#pragma unroll
  for (int it = 0; it < 2; ++it) {
    int cid = tid + it * 256;
    int r = cid >> 3, c0 = (cid & 7) << 3;
    *(v8h*)a_dst(Afrag, r, c0) = *(const v8h*)&q[(i0 + r) * HDIM + c0];
  }
  // stage k tile (32 keys x 64 dims): keys are d-contiguous -> K-contiguous for B
  {
    int n = tid >> 3, k0l = (tid & 7) << 3;
    *(v8h*)b_dst(Bfrag, n, k0l) = *(const v8h*)&k[(j0 + n) * HDIM + k0l];
  }
  __syncthreads();
  v8f acc = {};
  {
    v16h a0 = Afrag[0][wm][lane], b0 = Bfrag[0][wn][lane];
    v16h a1 = Afrag[1][wm][lane], b1 = Bfrag[1][wn][lane];
    acc = __builtin_amdgcn_wmma_f32_16x16x32_f16(false, a0, false, b0, (short)0, acc, false, false);
    acc = __builtin_amdgcn_wmma_f32_16x16x32_f16(false, a1, false, b1, (short)0, acc, false, false);
  }
  int nloc = lane & 15, mhalf = lane >> 4;
#pragma unroll
  for (int ii = 0; ii < 8; ++ii) {
    int i = i0 + wm * 16 + mhalf * 8 + ii;
    int j = j0 + wn * 16 + nloc;
    float raw = acc[ii];
    float den = fmaxf(qn[i], 1e-12f) * fmaxf(kn[j], 1e-12f);
    float cs = fminf(fmaxf(raw / den, -1.0f + 1e-7f), 1.0f - 1e-7f);
    float g = acosf(cs);
    if (MODE == 0) {
      int bin = (int)(g * ((float)NBINS / PI_F));
      bin = bin < 0 ? 0 : (bin > NBINS - 1 ? NBINS - 1 : bin);
      atomicAdd(&lhist[bin], 1u);
    } else {
      scores[i * SEQ + j] = raw * 0.125f;              // 1/sqrt(64)
      gmask[i * SEQ + j] = (g < thrb[0]) ? 1 : 0;
    }
  }
  if (MODE == 0) {
    __syncthreads();
    for (int t = tid; t < NBINS; t += 256)
      if (lhist[t]) atomicAdd(&histb[t], lhist[t]);
  }
}

// ---------------- K4: per-batch median threshold from histogram -------------------------
__global__ void median_kernel(const unsigned int* __restrict__ hist, float* __restrict__ thr) {
  if (threadIdx.x == 0) {
    for (int b = 0; b < 2; ++b) {
      unsigned long long total = 0ull;
      for (int i = 0; i < NBINS; ++i) total += hist[b * NBINS + i];
      unsigned long long half = total >> 1;
      unsigned long long c = 0ull; int bin = NBINS - 1;
      for (int i = 0; i < NBINS; ++i) {
        c += hist[b * NBINS + i];
        if (c >= half) { bin = i; break; }
      }
      thr[b] = PI_F * ((float)bin + 0.5f) / (float)NBINS;
    }
  }
}

// ---------------- K5: per-row top-k threshold (binary search, one wave per row) ---------
__global__ void topk_kernel(const float* __restrict__ scores, float* __restrict__ rowthr) {
  int lane = threadIdx.x & 31, wave = threadIdx.x >> 5;
  int row = blockIdx.x * 8 + wave;
  const float* rp = scores + row * SEQ;
  float vals[32];
  float vmx = -3.0e30f, vmn = 3.0e30f;
#pragma unroll
  for (int t = 0; t < 32; ++t) {
    float v = rp[lane + t * 32];
    vals[t] = v; vmx = fmaxf(vmx, v); vmn = fminf(vmn, v);
  }
  vmx = wave_max(vmx); vmn = wave_min(vmn);
  float lo = vmn, hi = vmx;
  for (int it = 0; it < 24; ++it) {
    float mid = 0.5f * (lo + hi);
    int cnt = 0;
#pragma unroll
    for (int t = 0; t < 32; ++t) cnt += (vals[t] >= mid) ? 1 : 0;
    cnt = wave_isum(cnt);
    if (cnt >= KSPARSE) lo = mid; else hi = mid;
  }
  if (lane == 0) rowthr[row] = lo;
}

// ---------------- K6: mask * scores -> softmax -> f16 attn (one block per row) ----------
__global__ void softmax_kernel(const float* __restrict__ scores, const unsigned char* __restrict__ gmask,
                               const float* __restrict__ rowthr, _Float16* __restrict__ attn) {
  __shared__ float red[8];
  __shared__ float bcast;
  int i = blockIdx.x;
  int tid = threadIdx.x;
  int lane = tid & 31, wave = tid >> 5;
  float t = rowthr[i];
  int rbi = i >> 6, rmi = i & 63;
  float mvals[4];
  float lmax = -3.0e30f;
#pragma unroll
  for (int u = 0; u < 4; ++u) {
    int j = tid + u * 256;
    float s = scores[i * SEQ + j];
    int rbj = j >> 6, rmj = j & 63;
    bool bm = (rbj == rbi) ||
              ((rmi < 8) && (rmj < 8) && (rbj == rbi - 1)) ||
              ((rmi < 8) && (rbj == rbi + 1));
    bool m = bm && (gmask[i * SEQ + j] != 0) && (s >= t);
    float ms = m ? s : 0.0f;     // reference multiplies scores by mask (zeros, not -inf)
    mvals[u] = ms;
    lmax = fmaxf(lmax, ms);
  }
  float wmx = wave_max(lmax);
  if (lane == 0) red[wave] = wmx;
  __syncthreads();
  if (wave == 0) {
    float v = (lane < 8) ? red[lane] : -3.0e30f;
    v = wave_max(v);
    if (lane == 0) bcast = v;
  }
  __syncthreads();
  float bmax = bcast;
  __syncthreads();
  float lsum = 0.f;
#pragma unroll
  for (int u = 0; u < 4; ++u) lsum += __expf(mvals[u] - bmax);
  float wsm = wave_sum(lsum);
  if (lane == 0) red[wave] = wsm;
  __syncthreads();
  if (wave == 0) {
    float v = (lane < 8) ? red[lane] : 0.f;
    v = wave_sum(v);
    if (lane == 0) bcast = v;
  }
  __syncthreads();
  float inv = 1.0f / bcast;
#pragma unroll
  for (int u = 0; u < 4; ++u) {
    int j = tid + u * 256;
    attn[i * SEQ + j] = (_Float16)(__expf(mvals[u] - bmax) * inv);
  }
}

// ---------------- K7: ctx = attn @ V^T-layout via WMMA ----------------------------------
// block tile 64(s) x 32(d); vT is [hd][S] so B staging is K-contiguous -> vectorized
__global__ void av_kernel(const _Float16* __restrict__ attn, const _Float16* __restrict__ vT,
                          _Float16* __restrict__ ctx, int bh) {
  __shared__ v16h Afrag[2][4][32];
  __shared__ v16h Bfrag[2][2][32];
  int tid = threadIdx.x;
  int lane = tid & 31, wave = tid >> 5;
  int wm = wave >> 1, wn = wave & 1;
  int i0 = blockIdx.x * 64, n0 = blockIdx.y * 32;
  v8f acc = {};
#pragma unroll 2
  for (int k0 = 0; k0 < SEQ; k0 += 64) {
#pragma unroll
    for (int it = 0; it < 2; ++it) {
      int cid = tid + it * 256;
      int r = cid >> 3, c0 = (cid & 7) << 3;
      *(v8h*)a_dst(Afrag, r, c0) = *(const v8h*)&attn[(i0 + r) * SEQ + k0 + c0];
    }
    {
      int n = tid >> 3, k0l = (tid & 7) << 3;
      *(v8h*)b_dst(Bfrag, n, k0l) = *(const v8h*)&vT[(n0 + n) * SEQ + k0 + k0l];
    }
    if (k0 + 64 < SEQ) __builtin_prefetch(&attn[i0 * SEQ + k0 + 64], 0, 1);
    __syncthreads();
    v16h a0 = Afrag[0][wm][lane], b0 = Bfrag[0][wn][lane];
    v16h a1 = Afrag[1][wm][lane], b1 = Bfrag[1][wn][lane];
    acc = __builtin_amdgcn_wmma_f32_16x16x32_f16(false, a0, false, b0, (short)0, acc, false, false);
    acc = __builtin_amdgcn_wmma_f32_16x16x32_f16(false, a1, false, b1, (short)0, acc, false, false);
    __syncthreads();
  }
  int batch = bh >> 3, h = bh & 7;
  int nloc = lane & 15, mhalf = lane >> 4;
#pragma unroll
  for (int ii = 0; ii < 8; ++ii) {
    int s = i0 + wm * 16 + mhalf * 8 + ii;
    int d = n0 + wn * 16 + nloc;
    ctx[((batch << 10) + s) * DMODEL + (h << 6) + d] = (_Float16)acc[ii];
  }
}

// ========================================================================================
extern "C" void kernel_launch(void* const* d_in, const int* in_sizes, int n_in,
                              void* d_out, int out_size, void* d_ws, size_t ws_size,
                              hipStream_t stream) {
  const float* x  = (const float*)d_in[0];
  const float* Wq = (const float*)d_in[1];
  const float* bq = (const float*)d_in[2];
  const float* Wk = (const float*)d_in[3];
  const float* bk = (const float*)d_in[4];
  const float* Wv = (const float*)d_in[5];
  const float* bv = (const float*)d_in[6];
  const float* Wo = (const float*)d_in[7];
  const float* bo = (const float*)d_in[8];

  char* ws = (char*)d_ws;
  size_t off = 0;
  auto alloc = [&](size_t bytes) -> void* {
    void* p = ws + off;
    off += (bytes + 255) & ~(size_t)255;
    return p;
  };
  const size_t BHS = 2 * NHEAD * SEQ;                 // 16384 rows
  _Float16* qh   = (_Float16*)alloc(BHS * HDIM * 2);  // 2 MB, [B,H,S,hd]
  _Float16* kh   = (_Float16*)alloc(BHS * HDIM * 2);  // 2 MB, [B,H,S,hd]
  _Float16* vTh  = (_Float16*)alloc(BHS * HDIM * 2);  // 2 MB, [B,H,hd,S] (transposed)
  _Float16* ctxh = (_Float16*)alloc((size_t)MROWS * DMODEL * 2); // 2 MB
  float* scores  = (float*)alloc((size_t)SEQ * SEQ * 4);         // 4 MB
  _Float16* attn = (_Float16*)alloc((size_t)SEQ * SEQ * 2);      // 2 MB
  unsigned char* gmask = (unsigned char*)alloc((size_t)SEQ * SEQ); // 1 MB
  float* qn      = (float*)alloc(BHS * 4);
  float* kn      = (float*)alloc(BHS * 4);
  unsigned int* hist = (unsigned int*)alloc(2 * NBINS * 4);
  float* thr     = (float*)alloc(64);
  float* rowthr  = (float*)alloc(SEQ * 4);

  dim3 blk(256);

  // Q/K/V projections (WMMA GEMMs); V written transposed
  gemm_wt_kernel<0, 0><<<dim3(MROWS / 64, DMODEL / 32), blk, 0, stream>>>(x, Wq, bq, qh, DMODEL, DMODEL);
  gemm_wt_kernel<0, 0><<<dim3(MROWS / 64, DMODEL / 32), blk, 0, stream>>>(x, Wk, bk, kh, DMODEL, DMODEL);
  gemm_wt_kernel<0, 2><<<dim3(MROWS / 64, DMODEL / 32), blk, 0, stream>>>(x, Wv, bv, vTh, DMODEL, DMODEL);

  // row norms for cosine similarity
  norms_kernel<<<dim3((unsigned)(BHS / 8)), blk, 0, stream>>>(qh, kh, qn, kn);

  // pass 1: geo-distance histogram per batch
  hipMemsetAsync(hist, 0, 2 * NBINS * 4, stream);
  for (int bh = 0; bh < 16; ++bh) {
    int b = bh >> 3;
    scores_kernel<0><<<dim3(SEQ / 64, SEQ / 32), blk, 0, stream>>>(
        qh + (size_t)bh * SEQ * HDIM, kh + (size_t)bh * SEQ * HDIM,
        qn + (size_t)bh * SEQ, kn + (size_t)bh * SEQ,
        scores, gmask, hist + b * NBINS, thr + b);
  }
  median_kernel<<<1, 64, 0, stream>>>(hist, thr);

  // pass 2 per head: scores + geo mask -> topk threshold -> softmax -> attn@V
  for (int bh = 0; bh < 16; ++bh) {
    int b = bh >> 3;
    scores_kernel<1><<<dim3(SEQ / 64, SEQ / 32), blk, 0, stream>>>(
        qh + (size_t)bh * SEQ * HDIM, kh + (size_t)bh * SEQ * HDIM,
        qn + (size_t)bh * SEQ, kn + (size_t)bh * SEQ,
        scores, gmask, hist + b * NBINS, thr + b);
    topk_kernel<<<dim3(SEQ / 8), blk, 0, stream>>>(scores, rowthr);
    softmax_kernel<<<dim3(SEQ), blk, 0, stream>>>(scores, gmask, rowthr, attn);
    av_kernel<<<dim3(SEQ / 64, HDIM / 32), blk, 0, stream>>>(
        attn, vTh + (size_t)bh * SEQ * HDIM, ctxh, bh);
  }

  // output projection -> f32 d_out [B,S,D]
  gemm_wt_kernel<1, 1><<<dim3(MROWS / 64, DMODEL / 32), blk, 0, stream>>>(ctxh, Wo, bo, d_out, DMODEL, DMODEL);
}
